// LSTMD_58548994179402
// MI455X (gfx1250) — compile-verified, matrix operand order checked by reading, not run on hardware
//
#include <hip/hip_runtime.h>
#include <hip/hip_bf16.h>

#define B_DIM 4096
#define T_DIM 128
#define D_DIM 64
#define H_DIM 128
#define G4    512   // 4*H
#define ROWS  16
#define NTHREADS 256

// swizzled-weight regions (bf16 element offsets inside workspace / LDS copy)
#define OFF_WGX   0        // 2 ktiles * 4 ntiles * 512
#define OFF_WGH   4096     // 2 * 8 * 512
#define OFF_WCAT  12288    // 6 * 32 * 512
#define OFF_WO1   110592   // 4 * 1 * 512
#define WSW_ELEMS 112640
#define WS_BIAS_BYTES (WSW_ELEMS * 2)

typedef __attribute__((ext_vector_type(16))) __bf16 v16bf;
typedef __attribute__((ext_vector_type(8)))  float  v8f;

static __device__ inline v8f wmma_bf16(v16bf a, v16bf b, v8f c) {
  return __builtin_amdgcn_wmma_f32_16x16x32_bf16(false, a, false, b, (short)0, c, false, false);
}
static __device__ inline float sigm(float x) { return 1.0f / (1.0f + __expf(-x)); }

#if __has_builtin(__builtin_amdgcn_tanhf)
static __device__ inline float fast_tanh(float x) { return __builtin_amdgcn_tanhf(x); }
#else
static __device__ inline float fast_tanh(float x) { return tanhf(x); }
#endif

// Scatter element (m,k) of a 16xK A-operand (bf16) into per-lane WMMA layout.
// Layout (ISA 7.12.2, 16-bit A 16x32): lane = m + 16*half, VGPR v holds K pair
// kbase = (v>=4?16:0) + half*8 + (v&3)*2.
static __device__ inline void scatter_a(__bf16* tile, int m, int k, __bf16 val) {
  int kt  = k >> 5, kp = k & 31;
  int t2  = kp >> 1, par = kp & 1;
  int half = (t2 >> 2) & 1;
  int v    = ((t2 >> 3) << 2) | (t2 & 3);
  tile[kt * 512 + (m + 16 * half) * 16 + v * 2 + par] = val;
}

// ---------------- prep: build bf16 swizzled weights + combined bias ----------------
__global__ void lstmd_prep(const float* __restrict__ w_gx, const float* __restrict__ w_gh,
                           const float* __restrict__ W_ih, const float* __restrict__ W_hh,
                           const float* __restrict__ b_ih, const float* __restrict__ b_hh,
                           const float* __restrict__ Wo1,
                           __bf16* __restrict__ sw, float* __restrict__ biasc) {
  int idx = blockIdx.x * blockDim.x + threadIdx.x;
  if (idx >= WSW_ELEMS) {
    int i = idx - WSW_ELEMS;
    if (i < G4) biasc[i] = b_ih[i] + b_hh[i];
    return;
  }
  int base, NT, region;
  if (idx < OFF_WGH)       { base = OFF_WGX;  NT = 4;  region = 0; }
  else if (idx < OFF_WCAT) { base = OFF_WGH;  NT = 8;  region = 1; }
  else if (idx < OFF_WO1)  { base = OFF_WCAT; NT = 32; region = 2; }
  else                     { base = OFF_WO1;  NT = 1;  region = 3; }
  int loc  = idx - base;
  int tile = loc >> 9, pos = loc & 511;
  int lane = pos >> 4, p = pos & 15;
  int nl   = lane & 15, half = lane >> 4;
  int v    = p >> 1,   par  = p & 1;
  int kl   = ((v >= 4) ? 16 : 0) + half * 8 + (v & 3) * 2 + par;
  int nt   = tile % NT, kt = tile / NT;
  int k    = kt * 32 + kl, n = nt * 16 + nl;
  float val;
  if (region == 0)      val = w_gx[k * D_DIM + n];                       // [D,D] K x N
  else if (region == 1) val = w_gh[k * H_DIM + n];                       // [D,H]
  else if (region == 2) val = (k < D_DIM) ? W_ih[n * D_DIM + k]          // W_ih^T
                                          : W_hh[n * H_DIM + (k - D_DIM)]; // W_hh^T
  else                  val = (n < 10) ? Wo1[n * H_DIM + k] : 0.0f;      // Wo1^T padded to 16
  sw[idx] = (__bf16)val;
}

// ---------------- main fused recurrent kernel ----------------
__global__ __launch_bounds__(NTHREADS, 1)
void lstmd_main(const float* __restrict__ x, const float* __restrict__ dt,
                const float* __restrict__ mask, const float* __restrict__ x_mean,
                const float* __restrict__ h0, const float* __restrict__ c0,
                const float* __restrict__ h_gx, const float* __restrict__ h_gh,
                const float* __restrict__ bo1, const float* __restrict__ Wo2,
                const float* __restrict__ bo2,
                const __bf16* __restrict__ sw, const float* __restrict__ biasc,
                float* __restrict__ outp, float* __restrict__ xhatp) {
  __shared__ __align__(16) __bf16 s_wsw[WSW_ELEMS];   // all swizzled weights (bf16)
  __shared__ __align__(16) __bf16 s_dtile[2 * 512];   // dt A-tile   [16 x 64]
  __shared__ __align__(16) __bf16 s_atile[6 * 512];   // gates A-tile [16 x 192] = xs || h
  __shared__ __align__(16) __bf16 s_h1t[4 * 512];     // head A-tile [16 x 128]
  __shared__ float s_bias[G4];
  __shared__ float s_gates[ROWS * G4];
  __shared__ float s_gx[ROWS * D_DIM];
  __shared__ float s_gh[ROWS * H_DIM];
  __shared__ float s_hid[ROWS * 16];
  __shared__ float s_sm[ROWS * 2];

  const int tid  = threadIdx.x;
  const int lane = tid & 31;
  const int wave = tid >> 5;
  const int b0   = blockIdx.x * ROWS;

  { // cooperative weight copy ws -> LDS (16B vectors)
    const uint4* src = (const uint4*)sw;
    uint4* dst = (uint4*)s_wsw;
    for (int i = tid; i < WSW_ELEMS / 8; i += NTHREADS) dst[i] = src[i];
    for (int i = tid; i < G4; i += NTHREADS) s_bias[i] = biasc[i];
  }

  // elementwise mappings: E1 = 16x64 (4/thread), E2 = 16x128 (8/thread)
  const int m1 = tid >> 4;
  const int k1 = (tid & 15) << 2;
  const int m2 = tid >> 4;
  const int c2 = (tid & 15) << 3;

  const int nlane = lane & 15;
  const int rbase = (lane >> 4) * 8;   // D-fragment row base

  // ---- hoist all loop-invariant global reads into registers ----
  float4 xmv = *(const float4*)(x_mean + k1);
  float xm[4] = {xmv.x, xmv.y, xmv.z, xmv.w};
  const float ghb = h_gh[wave * 16 + nlane];                       // gamma_h bias
  const float gxb = (wave < 4) ? h_gx[wave * 16 + nlane] : 0.0f;   // gamma_x bias
  const float b1  = (nlane < 10) ? bo1[nlane] : 0.0f;              // head L1 bias
  const int   hr  = lane >> 1, hp = lane & 1;                      // head L2 mapping
  float w2r[10];
#pragma unroll
  for (int u = 0; u < 10; ++u) w2r[u] = Wo2[hp * 10 + u];
  const float b2 = bo2[hp];

  float xp[4];
  float creg[8], hreg[8];
#pragma unroll
  for (int j = 0; j < 4; ++j) xp[j] = 0.0f;
#pragma unroll
  for (int j = 0; j < 8; ++j) { creg[j] = c0[c2 + j]; hreg[j] = h0[c2 + j]; }

  __syncthreads();

  for (int t = 0; t < T_DIM; ++t) {
    // ---- 1: load x/dt/mask tiles, scatter dt (bf16) ----
    size_t gbase = ((size_t)(b0 + m1) * T_DIM + t) * D_DIM + k1;
    float4 xv = *(const float4*)(x + gbase);
    float4 dv = *(const float4*)(dt + gbase);
    float4 mv = *(const float4*)(mask + gbase);
    float xr[4] = {xv.x, xv.y, xv.z, xv.w};
    float dr[4] = {dv.x, dv.y, dv.z, dv.w};
    float mr[4] = {mv.x, mv.y, mv.z, mv.w};
#pragma unroll
    for (int j = 0; j < 4; ++j) scatter_a(s_dtile, m1, k1 + j, (__bf16)dr[j]);
    if (t + 1 < T_DIM) {  // next-step tiles: re-read by this WGP next iteration
      __builtin_prefetch(x + gbase + D_DIM, 0, 3);
      __builtin_prefetch(dt + gbase + D_DIM, 0, 3);
      __builtin_prefetch(mask + gbase + D_DIM, 0, 3);
    }
    __syncthreads();

    // ---- 2: gamma GEMMs (bf16 WMMA, f32 acc) ----
    { // gamma_h: each wave owns one 16-col tile of H
      v8f acc = {0.f,0.f,0.f,0.f,0.f,0.f,0.f,0.f};
#pragma unroll
      for (int kt = 0; kt < 2; ++kt) {
        v16bf a = *(const v16bf*)(s_dtile + kt * 512 + lane * 16);
        v16bf b = *(const v16bf*)(s_wsw + OFF_WGH + (kt * 8 + wave) * 512 + lane * 16);
        acc = wmma_bf16(a, b, acc);
      }
      int col = wave * 16 + nlane;
#pragma unroll
      for (int v = 0; v < 8; ++v)
        s_gh[(rbase + v) * H_DIM + col] = __expf(-fmaxf(acc[v] + ghb, 0.f));
    }
    if (wave < 4) { // gamma_x
      v8f acc = {0.f,0.f,0.f,0.f,0.f,0.f,0.f,0.f};
#pragma unroll
      for (int kt = 0; kt < 2; ++kt) {
        v16bf a = *(const v16bf*)(s_dtile + kt * 512 + lane * 16);
        v16bf b = *(const v16bf*)(s_wsw + OFF_WGX + (kt * 4 + wave) * 512 + lane * 16);
        acc = wmma_bf16(a, b, acc);
      }
      int col = wave * 16 + nlane;
#pragma unroll
      for (int v = 0; v < 8; ++v)
        s_gx[(rbase + v) * D_DIM + col] = __expf(-fmaxf(acc[v] + gxb, 0.f));
    }
    __syncthreads();

    // ---- 3: xs imputation + state decay + A-tile assembly ----
    float xsv[4];
#pragma unroll
    for (int j = 0; j < 4; ++j) {
      float gx   = s_gx[m1 * D_DIM + k1 + j];
      float fill = (t == 0) ? xm[j] : (gx * xp[j] + (1.f - gx) * xm[j]);
      float xs   = xr[j] * mr[j] + (1.f - mr[j]) * fill;
      xp[j]  = xr[j];
      xsv[j] = xs;
      scatter_a(s_atile, m1, k1 + j, (__bf16)xs);
    }
    *(float4*)(xhatp + ((size_t)t * B_DIM + b0 + m1) * D_DIM + k1) =
        make_float4(xsv[0], xsv[1], xsv[2], xsv[3]);
#pragma unroll
    for (int j = 0; j < 8; ++j) {
      float gh = s_gh[m2 * H_DIM + c2 + j];
      float hd = gh * hreg[j];
      creg[j] *= gh;
      scatter_a(s_atile, m2, D_DIM + c2 + j, (__bf16)hd);
    }
    __syncthreads();

    // ---- 4: gates GEMM [16x192]x[192x512]: 4 n-tiles/wave, 6 k-steps ----
    {
      v8f acc0 = {0.f,0.f,0.f,0.f,0.f,0.f,0.f,0.f};
      v8f acc1 = acc0, acc2 = acc0, acc3 = acc0;
#pragma unroll
      for (int kt = 0; kt < 6; ++kt) {
        v16bf a = *(const v16bf*)(s_atile + kt * 512 + lane * 16);
        v16bf bq0 = *(const v16bf*)(s_wsw + OFF_WCAT + (kt * 32 + wave +  0) * 512 + lane * 16);
        v16bf bq1 = *(const v16bf*)(s_wsw + OFF_WCAT + (kt * 32 + wave +  8) * 512 + lane * 16);
        v16bf bq2 = *(const v16bf*)(s_wsw + OFF_WCAT + (kt * 32 + wave + 16) * 512 + lane * 16);
        v16bf bq3 = *(const v16bf*)(s_wsw + OFF_WCAT + (kt * 32 + wave + 24) * 512 + lane * 16);
        acc0 = wmma_bf16(a, bq0, acc0);
        acc1 = wmma_bf16(a, bq1, acc1);
        acc2 = wmma_bf16(a, bq2, acc2);
        acc3 = wmma_bf16(a, bq3, acc3);
      }
      v8f accs[4] = {acc0, acc1, acc2, acc3};
#pragma unroll
      for (int q = 0; q < 4; ++q) {
        int col = (wave + 8 * q) * 16 + nlane;
        float bb = s_bias[col];
#pragma unroll
        for (int v = 0; v < 8; ++v)
          s_gates[(rbase + v) * G4 + col] = accs[q][v] + bb;
      }
    }
    __syncthreads();

    // ---- 5: LSTM cell update (i,f,g,o) ----
#pragma unroll
    for (int j = 0; j < 8; ++j) {
      int col  = c2 + j;
      float ig = s_gates[m2 * G4 + col];
      float fg = s_gates[m2 * G4 + H_DIM + col];
      float gg = s_gates[m2 * G4 + 2 * H_DIM + col];
      float og = s_gates[m2 * G4 + 3 * H_DIM + col];
      float c1 = sigm(fg) * creg[j] + sigm(ig) * fast_tanh(gg);
      float h1 = sigm(og) * fast_tanh(c1);
      creg[j] = c1;
      hreg[j] = h1;
      scatter_a(s_h1t, m2, col, (__bf16)h1);
    }
    __syncthreads();

    // ---- 6: output head (wave 0): [16x128]x[128x16(pad of 10)] WMMA + tiny layer2 + softmax ----
    if (wave == 0) {
      v8f acc = {0.f,0.f,0.f,0.f,0.f,0.f,0.f,0.f};
#pragma unroll
      for (int kt = 0; kt < 4; ++kt) {
        v16bf a = *(const v16bf*)(s_h1t + kt * 512 + lane * 16);
        v16bf b = *(const v16bf*)(s_wsw + OFF_WO1 + kt * 512 + lane * 16);
        acc = wmma_bf16(a, b, acc);
      }
      if (nlane < 10) {
#pragma unroll
        for (int v = 0; v < 8; ++v)
          s_hid[(rbase + v) * 16 + nlane] = sigm(acc[v] + b1);
      }
      float s = b2;
#pragma unroll
      for (int u = 0; u < 10; ++u) s += w2r[u] * s_hid[hr * 16 + u];
      s = sigm(s);
      s_sm[hr * 2 + hp] = s;
      float s0 = s_sm[hr * 2 + 0], s1 = s_sm[hr * 2 + 1];
      float mx = fmaxf(s0, s1);
      float e0 = __expf(s0 - mx), e1 = __expf(s1 - mx);
      float o  = ((hp == 0) ? e0 : e1) / (e0 + e1);
      outp[((size_t)t * B_DIM + b0 + hr) * 2 + hp] = o;
    }
    __syncthreads();
  }
}

extern "C" void kernel_launch(void* const* d_in, const int* in_sizes, int n_in,
                              void* d_out, int out_size, void* d_ws, size_t ws_size,
                              hipStream_t stream) {
  const float* x      = (const float*)d_in[0];
  const float* dtp    = (const float*)d_in[1];
  const float* mask   = (const float*)d_in[2];
  /* d_in[3] = tp (int32, unused by forward) */
  const float* x_mean = (const float*)d_in[4];
  const float* h0     = (const float*)d_in[5];
  const float* c0     = (const float*)d_in[6];
  const float* w_gx   = (const float*)d_in[7];
  const float* h_gx   = (const float*)d_in[8];
  const float* w_gh   = (const float*)d_in[9];
  const float* h_gh   = (const float*)d_in[10];
  const float* W_ih   = (const float*)d_in[11];
  const float* W_hh   = (const float*)d_in[12];
  const float* b_ih   = (const float*)d_in[13];
  const float* b_hh   = (const float*)d_in[14];
  const float* Wo1    = (const float*)d_in[15];
  const float* bo1    = (const float*)d_in[16];
  const float* Wo2    = (const float*)d_in[17];
  const float* bo2    = (const float*)d_in[18];

  __bf16* sw   = (__bf16*)d_ws;
  float* biasc = (float*)((char*)d_ws + WS_BIAS_BYTES);

  float* outp  = (float*)d_out;                       // [T,B,2]
  float* xhatp = outp + (size_t)T_DIM * B_DIM * 2;    // [T,B,D]

  lstmd_prep<<<(WSW_ELEMS + G4 + NTHREADS - 1) / NTHREADS, NTHREADS, 0, stream>>>(
      w_gx, w_gh, W_ih, W_hh, b_ih, b_hh, Wo1, sw, biasc);

  lstmd_main<<<B_DIM / ROWS, NTHREADS, 0, stream>>>(
      x, dtp, mask, x_mean, h0, c0, h_gx, h_gh, bo1, Wo2, bo2, sw, biasc, outp, xhatp);
}
